// BiLSTM_CRF_50208167690393
// MI455X (gfx1250) — matrix operand order repaired and checked by reading, not compiled, not run on hardware
//
#include <hip/hip_runtime.h>
#include <hip/hip_bf16.h>

// ---------------- problem constants ----------------
static constexpr int S  = 512;
static constexpr int BB = 128;   // batch
static constexpr int E  = 128;   // embed dim (== H)
static constexpr int H  = 128;   // hidden per direction
static constexpr int G4 = 512;   // 4*H gates
static constexpr int NK = 32;    // CRF tags
static constexpr int NSTART = 30;
static constexpr int NEND   = 31;

typedef __attribute__((ext_vector_type(16))) _Float16 v16h;
typedef __attribute__((ext_vector_type(8)))  _Float16 v8h;
typedef __attribute__((ext_vector_type(8)))  float    v8f;

union V16HU { v16h v; v8h h[2]; };

// ---------------- WMMA helpers ----------------
__device__ __forceinline__ v8f wmma_f16(v16h a, v16h b, v8f c) {
  // D = A(16x32 f16) x B(32x16 f16) + C(16x16 f32)
  return __builtin_amdgcn_wmma_f32_16x16x32_f16(
      /*neg_a=*/false, a, /*neg_b=*/false, b,
      /*c_mod=*/(short)0, c, /*reuse_a=*/false, /*reuse_b=*/false);
}

// A-matrix 16x32 fragment from row-major [16+][ldk] f16 (base at row0,k0).
// Lanes 0-15: row=lane, K 0..15 ; lanes 16-31: row=lane-16, K 16..31.
__device__ __forceinline__ v16h load_a16(const _Float16* __restrict__ base,
                                         int ldk, int lane) {
  int row = lane & 15;
  int ko  = (lane >> 4) << 4;                 // 0 or 16
  const _Float16* p = base + row * ldk + ko;
  V16HU u;
  u.h[0] = *(const v8h*)(p);                  // K ko..ko+7
  u.h[1] = *(const v8h*)(p + 8);              // K ko+8..ko+15
  return u.v;
}

// B-matrix 32x16 fragment for B = W^T where W is row-major [N][ldk] f16.
// Lane n (<16): column n0+n, K k0..k0+15 ; lanes 16-31: K k0+16..k0+31.
__device__ __forceinline__ v16h load_bT16(const _Float16* __restrict__ W,
                                          int ldk, int n0, int k0, int lane) {
  int n  = lane & 15;
  int ko = (lane >> 4) << 4;                  // 0 or 16
  const _Float16* p = W + (size_t)(n0 + n) * ldk + k0 + ko;
  return *(const v16h*)(p);                   // 32 contiguous bytes
}

__device__ __forceinline__ float fsig(float x)  { return 1.0f / (1.0f + __expf(-x)); }
__device__ __forceinline__ float ftanh_(float x){ return 2.0f / (1.0f + __expf(-2.0f * x)) - 1.0f; }

// ---------------- small prep kernels ----------------
__global__ void cvt16_kernel(const float* __restrict__ src, _Float16* __restrict__ dst, int n) {
  int i = blockIdx.x * blockDim.x + threadIdx.x;
  if (i < n) dst[i] = (_Float16)src[i];
}

__global__ void bias_add_kernel(const float* __restrict__ a, const float* __restrict__ b,
                                float* __restrict__ out, int n) {
  int i = blockIdx.x * blockDim.x + threadIdx.x;
  if (i < n) out[i] = a[i] + b[i];
}

// X[(s*B+b)][e] = f16(embed_table[sentence[b][s]][e]);  grid = S*B blocks, 128 thr
__global__ void embed_kernel(const int* __restrict__ sentence,
                             const float* __restrict__ table,
                             _Float16* __restrict__ X) {
  int sb = blockIdx.x;
  int s = sb >> 7;           // / B (B=128)
  int b = sb & 127;
  int tok = sentence[b * S + s];
  X[(size_t)sb * E + threadIdx.x] = (_Float16)table[(size_t)tok * E + threadIdx.x];
}

// ---------------- input GEMM: Gpre[d][s*B+b][4H] = X @ Wih_d^T + bias_d ----------------
// wave handles one 16x64 output strip; 65536 waves total (2 dirs).
__global__ __launch_bounds__(256) void gemm_x_kernel(
    const _Float16* __restrict__ X,      // [S*B][E]
    const _Float16* __restrict__ Wih16,  // [2][512][128]
    const float* __restrict__ bias,      // [2][512]
    float* __restrict__ Gpre)            // [2][S*B][512]
{
  int wid  = blockIdx.x * 8 + (threadIdx.x >> 5);
  int lane = threadIdx.x & 31;
  int dir  = wid >> 15;
  int rem  = wid & 32767;
  int mt   = rem >> 3;                   // 0..4095  (16-row tile)
  int nq   = rem & 7;                    // 0..7     (64-col chunk)
  const _Float16* Wd = Wih16 + (size_t)dir * G4 * E;
  const float*    bd = bias  + dir * G4;
  float*          Gd = Gpre  + (size_t)dir * (S * BB) * G4;

  v16h a[4];
#pragma unroll
  for (int kb = 0; kb < 4; ++kb)
    a[kb] = load_a16(X + (size_t)mt * 16 * E + kb * 32, E, lane);

  int ccol  = lane & 15;
  int rbase = (lane >> 4) << 3;
#pragma unroll
  for (int nt = 0; nt < 4; ++nt) {
    int n0 = nq * 64 + nt * 16;
    v8f acc = {};
#pragma unroll
    for (int kb = 0; kb < 4; ++kb) {
      v16h bf = load_bT16(Wd, E, n0, kb * 32, lane);
      acc = wmma_f16(a[kb], bf, acc);
    }
    float bv = bd[n0 + ccol];
#pragma unroll
    for (int r = 0; r < 8; ++r) {
      int m = mt * 16 + rbase + r;
      Gd[(size_t)m * G4 + n0 + ccol] = acc[r] + bv;
    }
  }
}

// ---------------- persistent LSTM recurrence ----------------
// grid = 16 blocks (2 dirs x 8 batch-chunks of 16), 256 threads (8 waves).
// wave w owns hidden cols [16w,16w+16) for all 4 gates; cell state in VGPRs.
// Next timestep's Gpre tile (32 KB = 256 x 128B) is prefetched (one 128B chunk
// per thread) while the current step's WMMAs/activations run, hiding HBM
// latency on the sequential dependency chain.
__global__ __launch_bounds__(256) void lstm_kernel(
    const float* __restrict__ Gpre,      // [2][S*B][512]
    const _Float16* __restrict__ Whh16,  // [2][512][128]
    const float* __restrict__ h0,        // [2][B][H]
    const float* __restrict__ c0,        // [2][B][H]
    _Float16* __restrict__ hseq)         // [2][S*B][H]
{
  __shared__ __align__(32) _Float16 sH[16 * H];   // h_{t-1} for this batch chunk

  int blk = blockIdx.x;
  int dir = blk >> 3;
  int b0  = (blk & 7) * 16;
  int tid = threadIdx.x;
  int wv  = tid >> 5, lane = tid & 31;
  int n0  = wv * 16;

  const _Float16* W  = Whh16 + (size_t)dir * G4 * H;
  const float*    Gd = Gpre  + (size_t)dir * (S * BB) * G4;
  _Float16*       Hd = hseq  + (size_t)dir * (S * BB) * H;

  int ccol  = lane & 15;
  int rbase = (lane >> 4) << 3;

  // per-thread 128B prefetch slot within the 16x512 f32 step tile
  int pf_row = tid >> 4;          // 0..15
  int pf_col = (tid & 15) * 32;   // 0,32,...,480  (32 floats = 128 B)

  float c[8];
#pragma unroll
  for (int r = 0; r < 8; ++r) {
    int row = rbase + r;
    c[r] = c0[((size_t)dir * BB + b0 + row) * H + n0 + ccol];
  }
  for (int i = tid; i < 16 * H; i += 256) {
    int row = i >> 7, col = i & 127;
    sH[i] = (_Float16)h0[((size_t)dir * BB + b0 + row) * H + col];
  }
  // prefetch first step's tile
  {
    int t0 = dir ? (S - 1) : 0;
    __builtin_prefetch(&Gd[((size_t)t0 * BB + b0 + pf_row) * G4 + pf_col], 0, 1);
  }
  __syncthreads();

  for (int s = 0; s < S; ++s) {
    int t = dir ? (S - 1 - s) : s;

    v8f acc[4];
#pragma unroll
    for (int g = 0; g < 4; ++g) {
      int colg = g * H + n0 + ccol;
#pragma unroll
      for (int r = 0; r < 8; ++r) {
        int m = b0 + rbase + r;
        acc[g][r] = Gd[((size_t)t * BB + m) * G4 + colg];
      }
    }

    // prefetch next step's Gpre tile while this step computes
    if (s + 1 < S) {
      int tn = dir ? (t - 1) : (t + 1);
      __builtin_prefetch(&Gd[((size_t)tn * BB + b0 + pf_row) * G4 + pf_col], 0, 1);
    }

#pragma unroll
    for (int kb = 0; kb < 4; ++kb) {
      v16h a = load_a16(sH + kb * 32, H, lane);
#pragma unroll
      for (int g = 0; g < 4; ++g) {
        v16h bf = load_bT16(W, H, g * H + n0, kb * 32, lane);
        acc[g] = wmma_f16(a, bf, acc[g]);
      }
    }

    _Float16 hh[8];
#pragma unroll
    for (int r = 0; r < 8; ++r) {
      float ig = fsig(acc[0][r]);
      float fg = fsig(acc[1][r]);
      float gg = ftanh_(acc[2][r]);
      float og = fsig(acc[3][r]);
      c[r] = fg * c[r] + ig * gg;
      hh[r] = (_Float16)(og * ftanh_(c[r]));
    }

    __syncthreads();                       // everyone done reading old sH
#pragma unroll
    for (int r = 0; r < 8; ++r) {
      int row = rbase + r;
      sH[row * H + n0 + ccol] = hh[r];
      Hd[((size_t)t * BB + b0 + row) * H + n0 + ccol] = hh[r];
    }
    __syncthreads();                       // new h visible for next step
  }
}

// ---------------- output projection: feats[b][s][k] = [hf|hb] @ Wout^T + b_out ----------------
__global__ __launch_bounds__(256) void proj_kernel(
    const _Float16* __restrict__ hseq,   // [2][S*B][H]
    const _Float16* __restrict__ Wout16, // [32][256]
    const float* __restrict__ b_out,     // [32]
    float* __restrict__ feats)           // [B][S][32]
{
  int wid  = blockIdx.x * 8 + (threadIdx.x >> 5);   // M-tile 0..4095
  int lane = threadIdx.x & 31;
  const _Float16* hf = hseq;
  const _Float16* hb = hseq + (size_t)(S * BB) * H;

  v16h a[8];
#pragma unroll
  for (int kb = 0; kb < 8; ++kb) {
    int k0 = kb * 32;
    const _Float16* src = (k0 < H) ? (hf + (size_t)wid * 16 * H + k0)
                                   : (hb + (size_t)wid * 16 * H + (k0 - H));
    a[kb] = load_a16(src, H, lane);
  }

  int ccol  = lane & 15;
  int rbase = (lane >> 4) << 3;
#pragma unroll
  for (int nt = 0; nt < 2; ++nt) {
    v8f acc = {};
#pragma unroll
    for (int kb = 0; kb < 8; ++kb) {
      v16h bf = load_bT16(Wout16, 2 * H, nt * 16, kb * 32, lane);
      acc = wmma_f16(a[kb], bf, acc);
    }
    float bv = b_out[nt * 16 + ccol];
#pragma unroll
    for (int r = 0; r < 8; ++r) {
      int m = wid * 16 + rbase + r;     // = s*B + b
      int s = m >> 7;
      int b = m & 127;
      feats[(((size_t)b * S) + s) * NK + nt * 16 + ccol] = acc[r] + bv;
    }
  }
}

// ---------------- CRF numerator ----------------
__global__ void crf_num_kernel(const int* __restrict__ tags,
                               const float* __restrict__ feats,
                               const float* __restrict__ trans,
                               float* __restrict__ num) {
  __shared__ float T[NK * NK];
  int tid = threadIdx.x;                 // 128 threads, 1 block
  for (int i = tid; i < NK * NK; i += 128) T[i] = trans[i];
  __syncthreads();
  int b = tid;
  int prev = NSTART;
  float sum = 0.0f;
  for (int s = 0; s < S; ++s) {
    int tg = tags[b * S + s];
    sum += T[prev * NK + tg] + feats[((size_t)b * S + s) * NK + tg];
    prev = tg;
  }
  sum += T[prev * NK + NEND];
  num[b] = sum;
}

// ---------------- CRF denominator (forward algorithm) ----------------
// wave32 == 32 CRF states: lane = next-state; grid 16 x 256 (8 waves/block)
__global__ __launch_bounds__(256) void crf_den_kernel(
    const float* __restrict__ feats, const float* __restrict__ trans,
    float* __restrict__ den) {
  __shared__ float T[NK * NK];
  int tid = threadIdx.x;
  for (int i = tid; i < NK * NK; i += 256) T[i] = trans[i];
  __syncthreads();
  int wv = tid >> 5, lane = tid & 31;
  int b = blockIdx.x * 8 + wv;

  float alpha = (lane == NSTART) ? 0.0f : -10000.0f;
  for (int s = 0; s < S; ++s) {
    float m = -1e30f;
    for (int p = 0; p < NK; ++p) {
      float t = __shfl(alpha, p, 32) + T[lane * NK + p];   // transitions[next][prev]
      m = fmaxf(m, t);
    }
    float ssum = 0.0f;
    for (int p = 0; p < NK; ++p) {
      float t = __shfl(alpha, p, 32) + T[lane * NK + p];
      ssum += __expf(t - m);
    }
    alpha = m + __logf(ssum) + feats[((size_t)b * S + s) * NK + lane];
  }
  float v = alpha + T[lane * NK + NEND];
  float m = v;
  for (int o = 16; o > 0; o >>= 1) m = fmaxf(m, __shfl_xor(m, o, 32));
  float e = __expf(v - m);
  for (int o = 16; o > 0; o >>= 1) e += __shfl_xor(e, o, 32);
  if (lane == 0) den[b] = m + __logf(e);
}

// ---------------- final mean(num - den) ----------------
__global__ void final_kernel(const float* __restrict__ num,
                             const float* __restrict__ den,
                             float* __restrict__ out) {
  __shared__ float red[128];
  int tid = threadIdx.x;
  red[tid] = num[tid] - den[tid];
  __syncthreads();
  for (int off = 64; off > 0; off >>= 1) {
    if (tid < off) red[tid] += red[tid + off];
    __syncthreads();
  }
  if (tid == 0) out[0] = red[0] / 128.0f;
}

// ---------------- host launcher ----------------
extern "C" void kernel_launch(void* const* d_in, const int* in_sizes, int n_in,
                              void* d_out, int out_size, void* d_ws, size_t ws_size,
                              hipStream_t stream) {
  const int*   sentence = (const int*)d_in[0];
  const int*   tags     = (const int*)d_in[1];
  const float* table    = (const float*)d_in[2];
  const float* W_ih_f   = (const float*)d_in[3];
  const float* W_hh_f   = (const float*)d_in[4];
  const float* b_ih_f   = (const float*)d_in[5];
  const float* b_hh_f   = (const float*)d_in[6];
  const float* W_ih_b   = (const float*)d_in[7];
  const float* W_hh_b   = (const float*)d_in[8];
  const float* b_ih_b   = (const float*)d_in[9];
  const float* b_hh_b   = (const float*)d_in[10];
  const float* W_out    = (const float*)d_in[11];
  const float* b_out    = (const float*)d_in[12];
  const float* trans    = (const float*)d_in[13];
  const float* h0       = (const float*)d_in[14];
  const float* c0       = (const float*)d_in[15];
  float* out = (float*)d_out;

  // workspace partition (256B aligned)
  char* ws = (char*)d_ws;
  size_t off = 0;
  auto alloc = [&](size_t bytes) -> char* {
    char* p = ws + off;
    off = (off + bytes + 255) & ~(size_t)255;
    return p;
  };
  _Float16* Xf16   = (_Float16*)alloc((size_t)S * BB * E * 2);        // 16.8 MB
  _Float16* Wih16  = (_Float16*)alloc((size_t)2 * G4 * E * 2);        // 256 KB
  _Float16* Whh16  = (_Float16*)alloc((size_t)2 * G4 * H * 2);        // 256 KB
  _Float16* Wout16 = (_Float16*)alloc((size_t)NK * 2 * H * 2);        // 16 KB
  float*    bias   = (float*)alloc((size_t)2 * G4 * 4);               // 4 KB
  float*    Gpre   = (float*)alloc((size_t)2 * S * BB * G4 * 4);      // 268 MB
  _Float16* hseq   = (_Float16*)alloc((size_t)2 * S * BB * H * 2);    // 33.6 MB
  float*    feats  = (float*)alloc((size_t)BB * S * NK * 4);          // 8.4 MB
  float*    numv   = (float*)alloc(BB * 4);
  float*    denv   = (float*)alloc(BB * 4);

  // 1. weight conversion to f16 (+ folded biases)
  cvt16_kernel<<<(G4 * E + 255) / 256, 256, 0, stream>>>(W_ih_f, Wih16, G4 * E);
  cvt16_kernel<<<(G4 * E + 255) / 256, 256, 0, stream>>>(W_ih_b, Wih16 + G4 * E, G4 * E);
  cvt16_kernel<<<(G4 * H + 255) / 256, 256, 0, stream>>>(W_hh_f, Whh16, G4 * H);
  cvt16_kernel<<<(G4 * H + 255) / 256, 256, 0, stream>>>(W_hh_b, Whh16 + G4 * H, G4 * H);
  cvt16_kernel<<<(NK * 2 * H + 255) / 256, 256, 0, stream>>>(W_out, Wout16, NK * 2 * H);
  bias_add_kernel<<<2, 256, 0, stream>>>(b_ih_f, b_hh_f, bias, G4);
  bias_add_kernel<<<2, 256, 0, stream>>>(b_ih_b, b_hh_b, bias + G4, G4);

  // 2. embedding gather -> X f16 [S*B][E]
  embed_kernel<<<S * BB, E, 0, stream>>>(sentence, table, Xf16);

  // 3. big input GEMM (both directions): Gpre = X @ Wih^T + bias
  gemm_x_kernel<<<8192, 256, 0, stream>>>(Xf16, Wih16, bias, Gpre);

  // 4. sequential recurrence (16 persistent workgroups)
  lstm_kernel<<<16, 256, 0, stream>>>(Gpre, Whh16, h0, c0, hseq);

  // 5. output projection -> feats [B][S][K]
  proj_kernel<<<512, 256, 0, stream>>>(hseq, Wout16, b_out, feats);

  // 6. CRF score + partition function
  crf_num_kernel<<<1, 128, 0, stream>>>(tags, feats, trans, numv);
  crf_den_kernel<<<16, 256, 0, stream>>>(feats, trans, denv);

  // 7. mean(num - den)
  final_kernel<<<1, 128, 0, stream>>>(numv, denv, out);
}
